// TransformerDecoderOnlyBlock_47055661695523
// MI455X (gfx1250) — matrix operand order repaired and checked by reading, not compiled, not run on hardware
//
#include <hip/hip_runtime.h>
#include <stdint.h>

// ---------------------------------------------------------------------------
// Problem constants (fixed shapes from the reference)
// ---------------------------------------------------------------------------
#define BB    2
#define TT    2048
#define DD    1024
#define HH    16
#define DK    64
#define DFF   4096
#define NTOK  (BB * TT)          // 4096 rows
#define LN_EPS 1e-5f

// ---------------------------------------------------------------------------
// WMMA types (CDNA5 / gfx1250, wave32)
// ---------------------------------------------------------------------------
typedef __attribute__((ext_vector_type(16))) __bf16   v16bf;
typedef __attribute__((ext_vector_type(8)))  float    v8f;
typedef __attribute__((ext_vector_type(4)))  uint32_t v4u;

union BF16Frag {        // 8 VGPRs = 16 bf16 per lane
  v4u   q[2];
  v16bf v;
};

__device__ __forceinline__ unsigned short f2bf(float f) {
  uint32_t u = __float_as_uint(f);
  u += 0x7FFFu + ((u >> 16) & 1u);   // round-to-nearest-even
  return (unsigned short)(u >> 16);
}

__device__ __forceinline__ v8f wmma_bf16(v16bf a, v16bf b, v8f c) {
  return __builtin_amdgcn_wmma_f32_16x16x32_bf16(false, a, false, b,
                                                 (short)0, c, false, false);
}

// A fragment: 16x32 bf16, row-major source with leading dim lda.
// Layout used here: lanes 0-15 hold row M=lane, K=0..15;
//             lanes 16-31 hold row M=lane-16, K=16..31 (K-contiguous per lane).
__device__ __forceinline__ v16bf load_a_frag(const unsigned short* A, int lda,
                                             int row0, int k0, int lane) {
  const int r  = row0 + (lane & 15);
  const int kk = k0 + ((lane >> 4) << 4);
  BF16Frag f;
  f.q[0] = *(const v4u*)(A + (size_t)r * lda + kk);
  f.q[1] = *(const v4u*)(A + (size_t)r * lda + kk + 8);
  return f.v;
}

// B fragment: 32x16 bf16 (K x N), where column n of B = row n of row-major
// Wt(N x K).  Layout used here: lanes 0-15 hold K=0..15 of column N=lane,
// lanes 16-31 hold K=16..31 of column N=lane-16 (K-contiguous per lane).
__device__ __forceinline__ v16bf load_b_frag(const unsigned short* Wt, int ldw,
                                             int n0, int k0, int lane) {
  const int n  = n0 + (lane & 15);
  const int kk = k0 + ((lane >> 4) << 4);
  BF16Frag f;
  f.q[0] = *(const v4u*)(Wt + (size_t)n * ldw + kk);
  f.q[1] = *(const v4u*)(Wt + (size_t)n * ldw + kk + 8);
  return f.v;
}

// B fragment from per-head K (T x 64): B = K^T, column n = key n (clamped).
__device__ __forceinline__ v16bf load_k_frag(const unsigned short* Kp,
                                             int n0, int k0, int lane) {
  int n = n0 + (lane & 15);
  if (n > TT - 1) n = TT - 1;               // OOB keys are masked to 0 anyway
  const int kk = k0 + ((lane >> 4) << 4);
  BF16Frag f;
  f.q[0] = *(const v4u*)(Kp + (size_t)n * DK + kk);
  f.q[1] = *(const v4u*)(Kp + (size_t)n * DK + kk + 8);
  return f.v;
}

// B fragment from per-head V^T (64 x T): column n = dk index, K dim = key.
__device__ __forceinline__ v16bf load_vt_frag(const unsigned short* Vp,
                                              int k0, int n0, int lane) {
  const int n = n0 + (lane & 15);
  int kk = k0 + ((lane >> 4) << 4);
  if (kk > TT - 16) kk = TT - 16;           // clamp; masked P entries are 0
  BF16Frag f;
  f.q[0] = *(const v4u*)(Vp + (size_t)n * TT + kk);
  f.q[1] = *(const v4u*)(Vp + (size_t)n * TT + kk + 8);
  return f.v;
}

// ---------------------------------------------------------------------------
// f32 -> bf16 conversion
// ---------------------------------------------------------------------------
__global__ void cvt_f32_bf16(const float* __restrict__ src,
                             unsigned short* __restrict__ dst, size_t n) {
  size_t i = (size_t)blockIdx.x * blockDim.x + threadIdx.x;
  const size_t stride = (size_t)gridDim.x * blockDim.x;
  for (; i < n; i += stride) dst[i] = f2bf(src[i]);
}

// ---------------------------------------------------------------------------
// WMMA GEMM:  C[M,N] = A[M,K](bf16) * Wt[N,K](bf16)^T, f32 accum.
// 8 waves (2 in M, 4 in N); wave tile 64x32 (4 M-tiles x 2 N-tiles);
// workgroup tile 128x128.  Straight-line loads + WMMAs in the k-loop with
// unroll-2: the scheduler software-pipelines across iterations (loop-
// invariant base addresses + immediate offsets, staggered s_wait_loadcnt),
// as observed in the round-1 codegen.  ~44 FLOP per fragment byte keeps the
// L2-resident operand streams (entire 66MB working set fits the 192MB L2)
// below the L2 bandwidth ceiling at WMMA rates.
// Epilogue modes:
//   0: Q/K head scatter  -> outb[((b*H+h)*T + t)*64 + dk] = bf16(c*scale)
//   1: V head-transposed -> outb[((b*H+h)*64 + dk)*T + t] = bf16(c)
//   2: bias + exact GELU -> outb[row*N + col]
//   3: (+bias) + resid   -> outf[row*N + col] (f32)
// ---------------------------------------------------------------------------
__global__ __launch_bounds__(256)
void gemm_bf16_wmma(const unsigned short* __restrict__ A,
                    const unsigned short* __restrict__ Wt,
                    int M, int N, int K, int mode, float scale,
                    const float* __restrict__ bias,
                    const float* __restrict__ resid,
                    float* __restrict__ outf,
                    unsigned short* __restrict__ outb) {
  const int lane  = threadIdx.x & 31;
  const int wave  = threadIdx.x >> 5;
  const int waveM = wave >> 2;       // 0..1
  const int waveN = wave & 3;        // 0..3
  const int m_base = blockIdx.y * 128 + waveM * 64;
  const int n_base = blockIdx.x * 128 + waveN * 32;

  v8f acc[4][2] = {};

#pragma unroll 2
  for (int k0 = 0; k0 < K; k0 += 32) {
    v16bf a0 = load_a_frag(A, K, m_base,      k0, lane);
    v16bf a1 = load_a_frag(A, K, m_base + 16, k0, lane);
    v16bf a2 = load_a_frag(A, K, m_base + 32, k0, lane);
    v16bf a3 = load_a_frag(A, K, m_base + 48, k0, lane);
    v16bf b0 = load_b_frag(Wt, K, n_base,      k0, lane);
    v16bf b1 = load_b_frag(Wt, K, n_base + 16, k0, lane);
    acc[0][0] = wmma_bf16(a0, b0, acc[0][0]);
    acc[0][1] = wmma_bf16(a0, b1, acc[0][1]);
    acc[1][0] = wmma_bf16(a1, b0, acc[1][0]);
    acc[1][1] = wmma_bf16(a1, b1, acc[1][1]);
    acc[2][0] = wmma_bf16(a2, b0, acc[2][0]);
    acc[2][1] = wmma_bf16(a2, b1, acc[2][1]);
    acc[3][0] = wmma_bf16(a3, b0, acc[3][0]);
    acc[3][1] = wmma_bf16(a3, b1, acc[3][1]);
  }

  const int rsel = (lane >> 4) << 3;   // C layout: VGPR i -> row rsel+i
  const int csel = lane & 15;          //           col = lane & 15
#pragma unroll
  for (int mi = 0; mi < 4; ++mi) {
#pragma unroll
    for (int ni = 0; ni < 2; ++ni) {
#pragma unroll
      for (int i = 0; i < 8; ++i) {
        const int row = m_base + mi * 16 + rsel + i;
        const int col = n_base + ni * 16 + csel;
        float c = acc[mi][ni][i];
        if (mode == 0) {                       // Q/K: (B,H,T,DK) scatter
          const int bb = row >> 11, t = row & (TT - 1);
          const int h = col >> 6,  dk = col & 63;
          outb[(((size_t)(bb * HH + h) * TT + t) << 6) + dk] = f2bf(c * scale);
        } else if (mode == 1) {                // V: (B,H,DK,T) transposed
          const int bb = row >> 11, t = row & (TT - 1);
          const int h = col >> 6,  dk = col & 63;
          outb[(((size_t)(bb * HH + h) << 6) + dk) * TT + t] = f2bf(c);
        } else if (mode == 2) {                // bias + exact GELU -> bf16
          const float v = c + bias[col];
          const float g = 0.5f * v * (1.0f + erff(v * 0.70710678118654752f));
          outb[(size_t)row * N + col] = f2bf(g);
        } else {                               // (+bias) + residual -> f32
          float v = c + resid[(size_t)row * N + col];
          if (bias) v += bias[col];
          outf[(size_t)row * N + col] = v;
        }
      }
    }
  }
}

// ---------------------------------------------------------------------------
// Flash-style causal attention.  One wave owns 16 query rows; 8 waves per
// workgroup cover 128 rows; grid = (T/128, B*H).  dk = 64.
// S = Q*K^T via WMMA; online softmax with half-wave shfl reductions (the
// 16x16 C layout keeps each lane's 8 rows aligned with its O-accumulator
// rows, so rescale is lane-local); P goes C-layout -> A-layout through LDS.
// ---------------------------------------------------------------------------
__global__ __launch_bounds__(256)
void attn_flash(const unsigned short* __restrict__ Qh,
                const unsigned short* __restrict__ Kh,
                const unsigned short* __restrict__ VTh,
                unsigned short* __restrict__ attnb) {
  __shared__ unsigned short lds[8 * 16 * 40];   // per-wave 16x32 P tile, pad 40

  const int lane = threadIdx.x & 31;
  const int wave = threadIdx.x >> 5;
  const int bh = blockIdx.y;
  const int b = bh >> 4, h = bh & 15;
  const int q0 = blockIdx.x * 128 + wave * 16;

  const unsigned short* Qp = Qh  + (size_t)bh * TT * DK;
  const unsigned short* Kp = Kh  + (size_t)bh * TT * DK;
  const unsigned short* Vp = VTh + (size_t)bh * DK * TT;

  const v16bf aQ0 = load_a_frag(Qp, DK, q0, 0,  lane);
  const v16bf aQ1 = load_a_frag(Qp, DK, q0, 32, lane);

  const int rsel  = (lane >> 4) << 3;
  const int csel  = lane & 15;
  const int rbase = q0 + rsel;

  float m_i[8], l_i[8];
#pragma unroll
  for (int i = 0; i < 8; ++i) { m_i[i] = -1e30f; l_i[i] = 0.0f; }
  v8f o[4] = {};

  unsigned short* lp = lds + wave * 16 * 40;

  for (int j0 = 0; j0 <= q0 + 15; j0 += 32) {
    // ---- S tiles: cols [j0, j0+16), [j0+16, j0+32) ----
    v8f s0 = {}, s1 = {};
    {
      v16bf bk0 = load_k_frag(Kp, j0, 0, lane);
      v16bf bk1 = load_k_frag(Kp, j0, 32, lane);
      s0 = wmma_bf16(aQ0, bk0, s0);
      s0 = wmma_bf16(aQ1, bk1, s0);
    }
    {
      v16bf bk0 = load_k_frag(Kp, j0 + 16, 0, lane);
      v16bf bk1 = load_k_frag(Kp, j0 + 16, 32, lane);
      s1 = wmma_bf16(aQ0, bk0, s1);
      s1 = wmma_bf16(aQ1, bk1, s1);
    }
    // ---- causal mask (also covers clamped OOB keys) ----
    const int c0 = j0 + csel, c1 = j0 + 16 + csel;
#pragma unroll
    for (int i = 0; i < 8; ++i) {
      const int r = rbase + i;
      if (c0 > r) s0[i] = -1e30f;
      if (c1 > r) s1[i] = -1e30f;
    }
    // ---- online softmax (rows live in 16-lane halves) ----
    float alpha[8], p0[8], p1[8];
#pragma unroll
    for (int i = 0; i < 8; ++i) {
      float v = fmaxf(s0[i], s1[i]);
      v = fmaxf(v, __shfl_xor(v, 1, 32));
      v = fmaxf(v, __shfl_xor(v, 2, 32));
      v = fmaxf(v, __shfl_xor(v, 4, 32));
      v = fmaxf(v, __shfl_xor(v, 8, 32));
      const float mnew = fmaxf(m_i[i], v);
      alpha[i] = __expf(m_i[i] - mnew);
      p0[i] = __expf(s0[i] - mnew);
      p1[i] = __expf(s1[i] - mnew);
      float rs = p0[i] + p1[i];
      rs += __shfl_xor(rs, 1, 32);
      rs += __shfl_xor(rs, 2, 32);
      rs += __shfl_xor(rs, 4, 32);
      rs += __shfl_xor(rs, 8, 32);
      l_i[i] = l_i[i] * alpha[i] + rs;
      m_i[i] = mnew;
    }
#pragma unroll
    for (int nt = 0; nt < 4; ++nt)
#pragma unroll
      for (int i = 0; i < 8; ++i) o[nt][i] *= alpha[i];

    // ---- P: C layout -> LDS -> A-fragment layout ----
#pragma unroll
    for (int i = 0; i < 8; ++i) {
      const int r = rsel + i;
      lp[r * 40 + csel]      = f2bf(p0[i]);
      lp[r * 40 + 16 + csel] = f2bf(p1[i]);
    }
    asm volatile("s_wait_dscnt 0" ::: "memory");
    v16bf aP;
    {
      BF16Frag f;
      const unsigned short* base = lp + (lane & 15) * 40 + (rsel << 1);
      f.q[0] = *(const v4u*)(base);        // K chunk {0..7} / {16..23}
      f.q[1] = *(const v4u*)(base + 8);    // K chunk {8..15} / {24..31}
      aP = f.v;
    }
    // ---- O += P * V (V^T rows are K-contiguous) ----
#pragma unroll
    for (int nt = 0; nt < 4; ++nt) {
      v16bf bv = load_vt_frag(Vp, j0, nt * 16, lane);
      o[nt] = wmma_bf16(aP, bv, o[nt]);
    }
  }

  // ---- normalize and store back to (token, D) bf16 layout ----
  float inv[8];
#pragma unroll
  for (int i = 0; i < 8; ++i) inv[i] = 1.0f / l_i[i];
#pragma unroll
  for (int nt = 0; nt < 4; ++nt)
#pragma unroll
    for (int i = 0; i < 8; ++i) {
      const int t = rbase + i;
      const int col = (h << 6) + nt * 16 + csel;
      attnb[(((size_t)b * TT + t) << 10) + col] = f2bf(o[nt][i] * inv[i]);
    }
}

// ---------------------------------------------------------------------------
// Row LayerNorm over D=1024: one 256-thread block per row.
// Emits f32 (for residual / final output) and optionally bf16 (GEMM input).
// ---------------------------------------------------------------------------
__global__ __launch_bounds__(256)
void layernorm_rows(const float* __restrict__ Y,
                    const float* __restrict__ gamma,
                    const float* __restrict__ beta,
                    float* __restrict__ outf,
                    unsigned short* __restrict__ outb) {
  __shared__ float red[256];
  const int row = blockIdx.x;
  const float* y = Y + (size_t)row * DD;

  float v[4];
  float s = 0.0f;
#pragma unroll
  for (int j = 0; j < 4; ++j) { v[j] = y[threadIdx.x + 256 * j]; s += v[j]; }
  red[threadIdx.x] = s;
  __syncthreads();
  for (int off = 128; off > 0; off >>= 1) {
    if (threadIdx.x < off) red[threadIdx.x] += red[threadIdx.x + off];
    __syncthreads();
  }
  const float mean = red[0] * (1.0f / DD);
  __syncthreads();

  float sq = 0.0f;
#pragma unroll
  for (int j = 0; j < 4; ++j) { const float d = v[j] - mean; sq += d * d; }
  red[threadIdx.x] = sq;
  __syncthreads();
  for (int off = 128; off > 0; off >>= 1) {
    if (threadIdx.x < off) red[threadIdx.x] += red[threadIdx.x + off];
    __syncthreads();
  }
  const float rstd = rsqrtf(red[0] * (1.0f / DD) + LN_EPS);

#pragma unroll
  for (int j = 0; j < 4; ++j) {
    const int c = threadIdx.x + 256 * j;
    const float xo = (v[j] - mean) * rstd * gamma[c] + beta[c];
    if (outf) outf[(size_t)row * DD + c] = xo;
    if (outb) outb[(size_t)row * DD + c] = f2bf(xo);
  }
}

// ---------------------------------------------------------------------------
// Orchestration
// ---------------------------------------------------------------------------
extern "C" void kernel_launch(void* const* d_in, const int* in_sizes, int n_in,
                              void* d_out, int out_size, void* d_ws, size_t ws_size,
                              hipStream_t stream) {
  (void)in_sizes; (void)n_in; (void)out_size; (void)ws_size;

  const float* x  = (const float*)d_in[0];
  const float* Wq = (const float*)d_in[1];
  const float* Wk = (const float*)d_in[2];
  const float* Wv = (const float*)d_in[3];
  const float* Wo = (const float*)d_in[4];
  const float* W1 = (const float*)d_in[5];
  const float* b1 = (const float*)d_in[6];
  const float* W2 = (const float*)d_in[7];
  const float* b2 = (const float*)d_in[8];
  const float* W3 = (const float*)d_in[9];
  const float* b3 = (const float*)d_in[10];
  const float* gamma1 = (const float*)d_in[11];
  const float* beta1  = (const float*)d_in[12];
  const float* gamma2 = (const float*)d_in[13];
  const float* beta2  = (const float*)d_in[14];
  float* out = (float*)d_out;

  char* ws = (char*)d_ws;
  size_t off = 0;
  auto alloc = [&](size_t bytes) -> char* {
    char* p = ws + off;
    off = (off + bytes + 255) & ~(size_t)255;
    return p;
  };

  typedef unsigned short bf;
  bf* xb  = (bf*)alloc((size_t)NTOK * DD * 2);
  bf* wqb = (bf*)alloc((size_t)DD * DD * 2);
  bf* wkb = (bf*)alloc((size_t)DD * DD * 2);
  bf* wvb = (bf*)alloc((size_t)DD * DD * 2);
  bf* wob = (bf*)alloc((size_t)DD * DD * 2);
  bf* w1b = (bf*)alloc((size_t)DFF * DD * 2);
  bf* w2b = (bf*)alloc((size_t)DFF * DFF * 2);
  bf* w3b = (bf*)alloc((size_t)DD * DFF * 2);
  bf* qh  = (bf*)alloc((size_t)NTOK * DD * 2);   // (B,H,T,DK)
  bf* kh  = (bf*)alloc((size_t)NTOK * DD * 2);
  bf* vt  = (bf*)alloc((size_t)NTOK * DD * 2);   // (B,H,DK,T)
  bf* attnb = (bf*)alloc((size_t)NTOK * DD * 2);
  float* y1f = (float*)alloc((size_t)NTOK * DD * 4);
  float* x1f = (float*)alloc((size_t)NTOK * DD * 4);
  bf* x1b = (bf*)alloc((size_t)NTOK * DD * 2);
  bf* h2b = (bf*)alloc((size_t)NTOK * DFF * 2);
  // Aliases: qh..attnb (32MB) are dead once the Wo GEMM has consumed attnb,
  // and h1 is produced after that point.  y1f is dead after LN1.
  bf*    h1b = qh;
  float* y2f = y1f;

  const dim3 blk(256);

  // --- bf16 conversions ---
  cvt_f32_bf16<<<2048, blk, 0, stream>>>(x,  xb,  (size_t)NTOK * DD);
  cvt_f32_bf16<<<1024, blk, 0, stream>>>(Wq, wqb, (size_t)DD * DD);
  cvt_f32_bf16<<<1024, blk, 0, stream>>>(Wk, wkb, (size_t)DD * DD);
  cvt_f32_bf16<<<1024, blk, 0, stream>>>(Wv, wvb, (size_t)DD * DD);
  cvt_f32_bf16<<<1024, blk, 0, stream>>>(Wo, wob, (size_t)DD * DD);
  cvt_f32_bf16<<<2048, blk, 0, stream>>>(W1, w1b, (size_t)DFF * DD);
  cvt_f32_bf16<<<4096, blk, 0, stream>>>(W2, w2b, (size_t)DFF * DFF);
  cvt_f32_bf16<<<2048, blk, 0, stream>>>(W3, w3b, (size_t)DD * DFF);

  // --- QKV projections (Q pre-scaled by 1/sqrt(dk)) ---
  const dim3 gD(DD / 128, NTOK / 128);     // N=1024 -> (8, 32)
  const dim3 gF(DFF / 128, NTOK / 128);    // N=4096 -> (32, 32)
  gemm_bf16_wmma<<<gD, blk, 0, stream>>>(xb, wqb, NTOK, DD, DD, 0, 0.125f,
                                         nullptr, nullptr, nullptr, qh);
  gemm_bf16_wmma<<<gD, blk, 0, stream>>>(xb, wkb, NTOK, DD, DD, 0, 1.0f,
                                         nullptr, nullptr, nullptr, kh);
  gemm_bf16_wmma<<<gD, blk, 0, stream>>>(xb, wvb, NTOK, DD, DD, 1, 1.0f,
                                         nullptr, nullptr, nullptr, vt);

  // --- causal attention ---
  attn_flash<<<dim3(TT / 128, BB * HH), blk, 0, stream>>>(qh, kh, vt, attnb);

  // --- output projection + residual, LN1 ---
  gemm_bf16_wmma<<<gD, blk, 0, stream>>>(attnb, wob, NTOK, DD, DD, 3, 1.0f,
                                         nullptr, x, y1f, nullptr);
  layernorm_rows<<<NTOK, blk, 0, stream>>>(y1f, gamma1, beta1, x1f, x1b);

  // --- FFN: GELU(x1 W1^T + b1), GELU(h1 W2^T + b2), h2 W3^T + b3 + x1 ---
  gemm_bf16_wmma<<<gF, blk, 0, stream>>>(x1b, w1b, NTOK, DFF, DD, 2, 1.0f,
                                         b1, nullptr, nullptr, h1b);
  gemm_bf16_wmma<<<gF, blk, 0, stream>>>(h1b, w2b, NTOK, DFF, DFF, 2, 1.0f,
                                         b2, nullptr, nullptr, h2b);
  gemm_bf16_wmma<<<gD, blk, 0, stream>>>(h2b, w3b, NTOK, DD, DFF, 3, 1.0f,
                                         b3, x1f, y2f, nullptr);

  // --- LN2 -> final f32 output ---
  layernorm_rows<<<NTOK, blk, 0, stream>>>(y2f, gamma2, beta2, out, nullptr);
}
